// Attention_16896401343226
// MI455X (gfx1250) — matrix operand order repaired and checked by reading, not compiled, not run on hardware
//
#include <hip/hip_runtime.h>
#include <hip/hip_bf16.h>

typedef __attribute__((ext_vector_type(16))) _Float16 v16h;
typedef __attribute__((ext_vector_type(8)))  _Float16 v8h;
typedef __attribute__((ext_vector_type(4)))  _Float16 v4h;
typedef __attribute__((ext_vector_type(8)))  float    v8f;
typedef __attribute__((ext_vector_type(4)))  float    v4f;

#define B_   4
#define T_   8
#define C_   256
#define DH_  128
#define H_   64
#define W_   64
#define HW_  4096

// LDS halo tile: [4 rows][66 cols][32 ch], per-pixel stride padded 32 -> 40
// halves (20 words) so consecutive-pixel lanes hit distinct bank groups
// (20*i mod 64 cycles through 16 distinct starts; gcd(20,64)=4).
#define XSTRIDE 40

// ---------------------------------------------------------------------------
// Kernel 1: x (B*T, C, H, W) fp32 NCHW  ->  x16 (B*T, H, W, C) f16 NHWC
// LDS transpose: coalesced f32 reads along W, packed 16B f16 writes along C.
// grid = (HW/64, C/32, B*T), block = 256
// ---------------------------------------------------------------------------
__global__ __launch_bounds__(256) void convert_x_kernel(
    const float* __restrict__ x, _Float16* __restrict__ x16)
{
    __shared__ float Ts[64][33];           // [pixel][channel], padded
    const int pt = blockIdx.x;             // pixel tile (64 pixels)
    const int ct = blockIdx.y;             // channel tile (32 channels)
    const int bt = blockIdx.z;             // image
    const int tid = threadIdx.x;

    const float* src = x + ((size_t)bt * C_ + (size_t)ct * 32) * HW_ + (size_t)pt * 64;
    #pragma unroll
    for (int i = tid; i < 64 * 32; i += 256) {
        int cl = i >> 6, pl = i & 63;      // consecutive tid -> consecutive pixel
        Ts[pl][cl] = src[(size_t)cl * HW_ + pl];
    }
    __syncthreads();

    int pl = tid >> 2;
    int g  = (tid & 3) * 8;
    v8h o;
    #pragma unroll
    for (int j = 0; j < 8; ++j) o[j] = (_Float16)Ts[pl][g + j];
    *(v8h*)(x16 + ((size_t)bt * HW_ + (size_t)pt * 64 + pl) * C_ + ct * 32 + g) = o;
}

// ---------------------------------------------------------------------------
// Kernel 2: weight repack to f16, layout [oc][tap(=kh*3+kw)][cin]
//   which 0: w_q (128,256,3,3) -> wQ ; 1: w_k -> wK ; 2: w_v -> wV ;
//   which 3: w_out (256,128,3,3) -> wO
// ---------------------------------------------------------------------------
__global__ __launch_bounds__(256) void prep_weights_kernel(
    const float* __restrict__ wq, const float* __restrict__ wk,
    const float* __restrict__ wv, const float* __restrict__ wo,
    _Float16* __restrict__ wQ, _Float16* __restrict__ wK,
    _Float16* __restrict__ wV, _Float16* __restrict__ wO)
{
    const int PER = DH_ * 9 * C_;          // 294912 (same count for wO: 256*9*128)
    int idx = blockIdx.x * 256 + threadIdx.x;
    int which = idx / PER;
    int r = idx % PER;
    if (which < 3) {
        int c   = r % C_;
        int t2  = r / C_;
        int tap = t2 % 9;
        int oc  = t2 / 9;
        int kh = tap / 3, kw = tap % 3;
        const float* src = (which == 0) ? wq : ((which == 1) ? wk : wv);
        _Float16*    dst = (which == 0) ? wQ : ((which == 1) ? wK : wV);
        dst[r] = (_Float16)src[((oc * C_ + c) * 3 + kh) * 3 + kw];
    } else {
        int dh  = r % DH_;
        int t2  = r / DH_;
        int tap = t2 % 9;
        int oc  = t2 / 9;
        int kh = tap / 3, kw = tap % 3;
        wO[r] = (_Float16)wo[((oc * DH_ + dh) * 3 + kh) * 3 + kw];
    }
}

// ---------------------------------------------------------------------------
// WMMA implicit-GEMM conv core helpers (fragment fills in contiguous k = 16h + s order for A and B)
// ---------------------------------------------------------------------------
__device__ __forceinline__ void zero_acc(v8f acc[2][4]) {
    #pragma unroll
    for (int a = 0; a < 2; ++a)
        #pragma unroll
        for (int b = 0; b < 4; ++b)
            #pragma unroll
            for (int j = 0; j < 8; ++j) acc[a][b][j] = 0.0f;
}

__device__ __forceinline__ v16h ldsB(const _Float16* p) {
    v8h lo = *(const v8h*)(p);             // 16B -> ds_load_b128
    v8h hi = *(const v8h*)(p + 8);
    v16h b;
    #pragma unroll
    for (int j = 0; j < 8; ++j) { b[j] = lo[j]; b[8 + j] = hi[j]; }
    return b;
}

// ---------------------------------------------------------------------------
// Kernel 3: q/v/k convs as implicit GEMM, Cin=256, Cout=128.
//   grid.x = 32 pixel tiles (128 px = 2 rows), grid.y = job:
//   job 0..31  : q over image bt=job          -> q16  (NHWC f16)
//   job 32..63 : v over image bt=job-32       -> v16  (NHWC f16)
//   job 64..67 : k over image bt=(job-64)*T_  -> kbuf (NHWC f32)
// block = 256 threads = 8 waves, wave tile = 32 oc x 64 px
// ---------------------------------------------------------------------------
__global__ __launch_bounds__(256) void conv_qkv_kernel(
    const _Float16* __restrict__ x16,
    const _Float16* __restrict__ wQ, const _Float16* __restrict__ wV,
    const _Float16* __restrict__ wK,
    _Float16* __restrict__ q16, _Float16* __restrict__ v16b,
    float* __restrict__ kbuf)
{
    const int CIN = C_;                    // 256
    const int NCH = CIN / 32;              // 8 chunks
    __shared__ __align__(32) _Float16 Xs[4 * 66 * XSTRIDE];

    const int ntile = blockIdx.x;
    const int job   = blockIdx.y;
    const _Float16* w;
    int img, dimg;
    bool is_k = false;
    if (job < 32)      { w = wQ; img = job;          dimg = job;      }
    else if (job < 64) { w = wV; img = job - 32;     dimg = job - 32; }
    else               { w = wK; img = (job - 64) * T_; dimg = job - 64; is_k = true; }
    const _Float16* xin = x16 + (size_t)img * HW_ * CIN;

    const int tid  = threadIdx.x;
    const int lane = tid & 31;
    const int wid  = tid >> 5;
    const int wm   = wid >> 1;             // 0..3 -> 32-row band
    const int wn   = wid & 1;              // 0..1 -> 64-px band
    const int h0   = ntile * 2;

    v8f acc[2][4];
    zero_acc(acc);

    const int kbA = (lane >> 4) * 16;      // A-fragment K base (contiguous k = 16h + s, same order as B)
    const int kbB = (lane >> 4) * 16;      // B-fragment K base (contiguous k = 16h + s, same order as A)

    for (int ch = 0; ch < NCH; ++ch) {
        __syncthreads();
        // stage 4 rows x 66 cols x 32 ch halo tile (zero-padded) into LDS
        for (int t = tid; t < 4 * 66 * 4; t += 256) {
            int g  = t & 3;
            int rc = t >> 2;
            int cc = rc % 66;
            int r  = rc / 66;
            int h  = h0 - 1 + r;
            int wc = cc - 1;
            v8h val;
            #pragma unroll
            for (int j = 0; j < 8; ++j) val[j] = (_Float16)0.0f;
            if ((unsigned)h < (unsigned)H_ && (unsigned)wc < (unsigned)W_)
                val = *(const v8h*)(xin + (size_t)(h * W_ + wc) * CIN + ch * 32 + g * 8);
            *(v8h*)(Xs + (r * 66 + cc) * XSTRIDE + g * 8) = val;
        }
        __syncthreads();

        if (ch + 1 < NCH)   // speculative prefetch of next channel chunk
            __builtin_prefetch(xin + (size_t)(h0 * W_) * CIN + (ch + 1) * 32, 0, 1);

        const int kc = ch * 32;
        #pragma unroll
        for (int tap = 0; tap < 9; ++tap) {
            const int kh = tap / 3, kw = tap % 3;
            // A fragments: two 16B global loads per lane per M-subtile
            v16h afrag[2];
            #pragma unroll
            for (int ms = 0; ms < 2; ++ms) {
                int m = wm * 32 + ms * 16 + (lane & 15);
                const _Float16* wrow = w + (size_t)m * (9 * CIN) + tap * CIN + kc + kbA;
                v8h lo = *(const v8h*)(wrow);
                v8h hi = *(const v8h*)(wrow + 8);
                #pragma unroll
                for (int j = 0; j < 8; ++j) { afrag[ms][j] = lo[j]; afrag[ms][8 + j] = hi[j]; }
            }
            #pragma unroll
            for (int ns = 0; ns < 4; ++ns) {
                int pl = wn * 64 + ns * 16 + (lane & 15);
                int r  = (pl >> 6) + kh;
                int cc = (pl & 63) + kw;
                v16h bfrag = ldsB(Xs + (r * 66 + cc) * XSTRIDE + kbB);
                #pragma unroll
                for (int ms = 0; ms < 2; ++ms)
                    acc[ms][ns] = __builtin_amdgcn_wmma_f32_16x16x32_f16(
                        false, afrag[ms], false, bfrag, (short)0, acc[ms][ns], false, false);
            }
        }
    }

    // epilogue: D layout -> 8 consecutive oc per lane per acc
    const int ocb  = wm * 32 + (lane >> 4) * 8;
    const int pixb = ntile * 128 + wn * 64;
    if (!is_k) {
        _Float16* dbase = (job < 32 ? q16 : v16b) + (size_t)dimg * HW_ * DH_;
        #pragma unroll
        for (int ms = 0; ms < 2; ++ms)
            #pragma unroll
            for (int ns = 0; ns < 4; ++ns) {
                int oc  = ocb + ms * 16;
                int pix = pixb + ns * 16 + (lane & 15);
                v8h o;
                #pragma unroll
                for (int r = 0; r < 8; ++r) o[r] = (_Float16)acc[ms][ns][r];
                *(v8h*)(dbase + (size_t)pix * DH_ + oc) = o;
            }
    } else {
        float* dbase = kbuf + (size_t)dimg * HW_ * DH_;
        #pragma unroll
        for (int ms = 0; ms < 2; ++ms)
            #pragma unroll
            for (int ns = 0; ns < 4; ++ns) {
                int oc  = ocb + ms * 16;
                int pix = pixb + ns * 16 + (lane & 15);
                v4f lo, hi;
                #pragma unroll
                for (int r = 0; r < 4; ++r) { lo[r] = acc[ms][ns][r]; hi[r] = acc[ms][ns][4 + r]; }
                *(v4f*)(dbase + (size_t)pix * DH_ + oc)     = lo;
                *(v4f*)(dbase + (size_t)pix * DH_ + oc + 4) = hi;
            }
    }
}

// ---------------------------------------------------------------------------
// Kernel 4: per-pixel attention. One wave per (b, pixel); lane owns 4 dh.
// dots in fp32, softmax over T=8, pooled written as f16 NHWC.
// grid = B*HW/8 blocks of 256
// ---------------------------------------------------------------------------
__global__ __launch_bounds__(256) void attn_kernel(
    const _Float16* __restrict__ q16, const _Float16* __restrict__ v16b,
    const float* __restrict__ kbuf, _Float16* __restrict__ pooled16)
{
    const int wid  = threadIdx.x >> 5;
    const int lane = threadIdx.x & 31;
    const int job  = blockIdx.x * 8 + wid;     // 0 .. B*HW-1
    const int b    = job >> 12;
    const int pix  = job & (HW_ - 1);

    v4f kv = *(const v4f*)(kbuf + ((size_t)b * HW_ + pix) * DH_ + lane * 4);

    float dots[T_];
    #pragma unroll
    for (int t = 0; t < T_; ++t) {
        const _Float16* qp = q16 + (((size_t)(b * T_ + t) * HW_ + pix) * DH_) + lane * 4;
        v4h qv = *(const v4h*)qp;
        float s = (float)qv[0] * kv[0] + (float)qv[1] * kv[1] +
                  (float)qv[2] * kv[2] + (float)qv[3] * kv[3];
        #pragma unroll
        for (int off = 16; off > 0; off >>= 1) s += __shfl_xor(s, off, 32);
        dots[t] = s;
    }
    float mx = dots[0];
    #pragma unroll
    for (int t = 1; t < T_; ++t) mx = fmaxf(mx, dots[t]);
    float e[T_], sum = 0.0f;
    #pragma unroll
    for (int t = 0; t < T_; ++t) { e[t] = __expf(dots[t] - mx); sum += e[t]; }
    float inv = 1.0f / sum;

    float p0 = 0.f, p1 = 0.f, p2 = 0.f, p3 = 0.f;
    #pragma unroll
    for (int t = 0; t < T_; ++t) {
        const _Float16* vp = v16b + (((size_t)(b * T_ + t) * HW_ + pix) * DH_) + lane * 4;
        v4h vv = *(const v4h*)vp;
        float a = e[t] * inv;
        p0 += a * (float)vv[0]; p1 += a * (float)vv[1];
        p2 += a * (float)vv[2]; p3 += a * (float)vv[3];
    }
    v4h o; o[0] = (_Float16)p0; o[1] = (_Float16)p1; o[2] = (_Float16)p2; o[3] = (_Float16)p3;
    *(v4h*)(pooled16 + ((size_t)b * HW_ + pix) * DH_ + lane * 4) = o;
}

// ---------------------------------------------------------------------------
// Kernel 5: output conv, Cin=128 (4 chunks), Cout=256, + bias -> outc f32 NCHW
// grid = (32 pixel tiles, 2 M tiles, B), block = 256
// ---------------------------------------------------------------------------
__global__ __launch_bounds__(256) void conv_out_kernel(
    const _Float16* __restrict__ pooled16, const _Float16* __restrict__ wO,
    const float* __restrict__ b_out, float* __restrict__ outc)
{
    const int CIN = DH_;                   // 128
    const int NCH = CIN / 32;              // 4 chunks
    __shared__ __align__(32) _Float16 Xs[4 * 66 * XSTRIDE];

    const int ntile = blockIdx.x;
    const int mtile = blockIdx.y;          // 0..1 -> which 128 of 256 oc
    const int bimg  = blockIdx.z;
    const _Float16* xin = pooled16 + (size_t)bimg * HW_ * CIN;

    const int tid  = threadIdx.x;
    const int lane = tid & 31;
    const int wid  = tid >> 5;
    const int wm   = wid >> 1;
    const int wn   = wid & 1;
    const int h0   = ntile * 2;

    v8f acc[2][4];
    zero_acc(acc);

    const int kbA = (lane >> 4) * 16;
    const int kbB = (lane >> 4) * 16;

    for (int ch = 0; ch < NCH; ++ch) {
        __syncthreads();
        for (int t = tid; t < 4 * 66 * 4; t += 256) {
            int g  = t & 3;
            int rc = t >> 2;
            int cc = rc % 66;
            int r  = rc / 66;
            int h  = h0 - 1 + r;
            int wc = cc - 1;
            v8h val;
            #pragma unroll
            for (int j = 0; j < 8; ++j) val[j] = (_Float16)0.0f;
            if ((unsigned)h < (unsigned)H_ && (unsigned)wc < (unsigned)W_)
                val = *(const v8h*)(xin + (size_t)(h * W_ + wc) * CIN + ch * 32 + g * 8);
            *(v8h*)(Xs + (r * 66 + cc) * XSTRIDE + g * 8) = val;
        }
        __syncthreads();

        const int kc = ch * 32;
        #pragma unroll
        for (int tap = 0; tap < 9; ++tap) {
            const int kh = tap / 3, kw = tap % 3;
            v16h afrag[2];
            #pragma unroll
            for (int ms = 0; ms < 2; ++ms) {
                int m = mtile * 128 + wm * 32 + ms * 16 + (lane & 15);
                const _Float16* wrow = wO + (size_t)m * (9 * CIN) + tap * CIN + kc + kbA;
                v8h lo = *(const v8h*)(wrow);
                v8h hi = *(const v8h*)(wrow + 8);
                #pragma unroll
                for (int j = 0; j < 8; ++j) { afrag[ms][j] = lo[j]; afrag[ms][8 + j] = hi[j]; }
            }
            #pragma unroll
            for (int ns = 0; ns < 4; ++ns) {
                int pl = wn * 64 + ns * 16 + (lane & 15);
                int r  = (pl >> 6) + kh;
                int cc = (pl & 63) + kw;
                v16h bfrag = ldsB(Xs + (r * 66 + cc) * XSTRIDE + kbB);
                #pragma unroll
                for (int ms = 0; ms < 2; ++ms)
                    acc[ms][ns] = __builtin_amdgcn_wmma_f32_16x16x32_f16(
                        false, afrag[ms], false, bfrag, (short)0, acc[ms][ns], false, false);
            }
        }
    }

    // epilogue: + bias, f32 NCHW
    const int pixb = ntile * 128 + wn * 64;
    #pragma unroll
    for (int ms = 0; ms < 2; ++ms)
        #pragma unroll
        for (int ns = 0; ns < 4; ++ns) {
            int ocb = mtile * 128 + wm * 32 + ms * 16 + (lane >> 4) * 8;
            int pix = pixb + ns * 16 + (lane & 15);
            #pragma unroll
            for (int r = 0; r < 8; ++r) {
                int oc = ocb + r;
                outc[((size_t)bimg * C_ + oc) * HW_ + pix] = acc[ms][ns][r] + b_out[oc];
            }
        }
}

// ---------------------------------------------------------------------------
// Kernel 6: broadcast outc (B,C,H,W) over T into d_out (B,T,C,H,W), float4
// ---------------------------------------------------------------------------
__global__ __launch_bounds__(256) void bcast_kernel(
    const float* __restrict__ outc, float* __restrict__ out)
{
    const size_t PER = (size_t)C_ * HW_ / 4;               // float4 per (b,t)
    size_t i   = (size_t)blockIdx.x * 256 + threadIdx.x;   // float4 index into out
    size_t bt  = i / PER;
    size_t rem = i % PER;
    size_t b   = bt / T_;
    v4f val = *(const v4f*)(outc + (b * PER + rem) * 4);
    *(v4f*)(out + i * 4) = val;
}

// ---------------------------------------------------------------------------
extern "C" void kernel_launch(void* const* d_in, const int* in_sizes, int n_in,
                              void* d_out, int out_size, void* d_ws, size_t ws_size,
                              hipStream_t stream) {
    const float* x    = (const float*)d_in[0];
    const float* w_k  = (const float*)d_in[1];
    const float* w_q  = (const float*)d_in[2];
    const float* w_v  = (const float*)d_in[3];
    const float* w_o  = (const float*)d_in[4];
    const float* b_o  = (const float*)d_in[5];
    float* out = (float*)d_out;

    char* ws = (char*)d_ws;
    size_t off = 0;
    auto take = [&](size_t bytes) -> char* {
        char* p = ws + off;
        off += (bytes + 255) & ~(size_t)255;
        return p;
    };
    _Float16* x16  = (_Float16*)take((size_t)B_ * T_ * HW_ * C_ * 2);   // 64 MB
    _Float16* wQ   = (_Float16*)take((size_t)DH_ * 9 * C_ * 2);
    _Float16* wK   = (_Float16*)take((size_t)DH_ * 9 * C_ * 2);
    _Float16* wV   = (_Float16*)take((size_t)DH_ * 9 * C_ * 2);
    _Float16* wO   = (_Float16*)take((size_t)C_ * 9 * DH_ * 2);
    _Float16* q16  = (_Float16*)take((size_t)B_ * T_ * HW_ * DH_ * 2);  // 32 MB
    _Float16* v16b = (_Float16*)take((size_t)B_ * T_ * HW_ * DH_ * 2);  // 32 MB
    float*    kbuf = (float*)   take((size_t)B_ * HW_ * DH_ * 4);       //  8 MB
    _Float16* pool = (_Float16*)take((size_t)B_ * HW_ * DH_ * 2);       //  4 MB
    float*    outc = (float*)   take((size_t)B_ * C_ * HW_ * 4);        // 16 MB

    // 1) layout conversions
    convert_x_kernel<<<dim3(HW_ / 64, C_ / 32, B_ * T_), 256, 0, stream>>>(x, x16);
    prep_weights_kernel<<<(4 * DH_ * 9 * C_) / 256, 256, 0, stream>>>(
        w_q, w_k, w_v, w_o, wQ, wK, wV, wO);

    // 2) q, v, k convs (implicit GEMM WMMA)
    conv_qkv_kernel<<<dim3(HW_ / 128, 68, 1), 256, 0, stream>>>(
        x16, wQ, wV, wK, q16, v16b, kbuf);

    // 3) per-pixel temporal attention
    attn_kernel<<<(B_ * HW_) / 8, 256, 0, stream>>>(q16, v16b, kbuf, pool);

    // 4) output conv + bias
    conv_out_kernel<<<dim3(HW_ / 128, 2, B_), 256, 0, stream>>>(pool, wO, b_o, outc);

    // 5) broadcast over T into d_out
    bcast_kernel<<<(unsigned)(((size_t)B_ * T_ * C_ * HW_ / 4) / 256), 256, 0, stream>>>(
        outc, out);
}
